// Compled_reco_79353815761562
// MI455X (gfx1250) — matrix operand order repaired and checked by reading, not compiled, not run on hardware
//
#include <hip/hip_runtime.h>

typedef __attribute__((ext_vector_type(16))) _Float16 v16h;
typedef __attribute__((ext_vector_type(8)))  float    v8f;

#define EPSF     1e-8f
#define INV_T    2.0f               // 1 / TEMP (TEMP = 0.5)
#define SQRT257  16.03121954188140f // sqrt(K+1) for bg candidate-axis norm

// Hardware transcendental approximations (v_exp_f32/v_log_f32 are base-2).
__device__ __forceinline__ float fast_exp(float x) {
  return __builtin_amdgcn_exp2f(x * 1.4426950408889634f);
}
__device__ __forceinline__ float fast_log(float x) {
  return __builtin_amdgcn_logf(x) * 0.6931471805599453f;
}
__device__ __forceinline__ float fast_rcp(float x) { return __builtin_amdgcn_rcpf(x); }
__device__ __forceinline__ float fast_rsq(float x) { return __builtin_amdgcn_rsqf(x); }

__device__ __forceinline__ float wave_sum32(float v) {
#pragma unroll
  for (int off = 16; off >= 1; off >>= 1)
    v += __shfl_xor(v, off, 32);
  return v;
}

// One wave32 per (branch, c, q) task. 4096 tasks = 512 blocks x 8 waves.
__global__ __launch_bounds__(256) void contrast_kernel(
    const float* __restrict__ fore_rep,   // (131072, 128)
    const float* __restrict__ proto,      // (9, 128)
    const float* __restrict__ memo,       // (100000, 128)
    const int*   __restrict__ anchor_idx, // (8, 256)
    const int*   __restrict__ nfg,        // (8, 256, 256)
    const int*   __restrict__ nbg,        // (8, 256, 256)
    float*       __restrict__ ws)         // 4096 CE values: [fg 2048][bg 2048]
{
  const int lane   = threadIdx.x & 31;
  const int task   = blockIdx.x * 8 + (threadIdx.x >> 5); // 0..4095
  const int branch = task >> 11;                          // 0 = fg, 1 = bg
  const int pair   = task & 2047;                         // c*256 + q
  const int c      = pair >> 8;

  const long  aidx = anchor_idx[pair];
  const float* __restrict__ arow = fore_rep + aidx * 128;
  const float* __restrict__ prow = proto + (long)(c + 1) * 128;

  if (branch == 0) {
    // ================= FG: 257 cosine dots via WMMA =================
    // anchor L2 norm (f32); 1/max(sqrt(ss),1e-8) == rsq(max(ss,1e-16))
    float4 av  = ((const float4*)arow)[lane];
    float  ssa = av.x*av.x + av.y*av.y + av.z*av.z + av.w*av.w;
    ssa = wave_sum32(ssa);
    const float sa = fast_rsq(fmaxf(ssa, 1e-16f));

    // B = normalized anchor replicated into all 16 columns.
    // B k order used here: lanes 0-15 hold K=0..15, lanes 16-31 hold K=16..31.
    v16h B[4];
#pragma unroll
    for (int ch = 0; ch < 4; ++ch) {
      const float* bp = arow + ch * 32 + ((lane < 16) ? 0 : 16);
      float4 x0 = ((const float4*)bp)[0];
      float4 x1 = ((const float4*)bp)[1];
      float4 x2 = ((const float4*)bp)[2];
      float4 x3 = ((const float4*)bp)[3];
      B[ch][0]=(_Float16)(x0.x*sa);  B[ch][1]=(_Float16)(x0.y*sa);
      B[ch][2]=(_Float16)(x0.z*sa);  B[ch][3]=(_Float16)(x0.w*sa);
      B[ch][4]=(_Float16)(x1.x*sa);  B[ch][5]=(_Float16)(x1.y*sa);
      B[ch][6]=(_Float16)(x1.z*sa);  B[ch][7]=(_Float16)(x1.w*sa);
      B[ch][8]=(_Float16)(x2.x*sa);  B[ch][9]=(_Float16)(x2.y*sa);
      B[ch][10]=(_Float16)(x2.z*sa); B[ch][11]=(_Float16)(x2.w*sa);
      B[ch][12]=(_Float16)(x3.x*sa); B[ch][13]=(_Float16)(x3.y*sa);
      B[ch][14]=(_Float16)(x3.z*sa); B[ch][15]=(_Float16)(x3.w*sa);
    }

    const int* __restrict__ nrow = nfg + (long)pair * 256;
    const int r     = lane & 15;
    const int kbase = (lane < 16) ? 0 : 16; // A-layout K base within a 32-chunk
    const int mbase = (lane < 16) ? 0 : 8; // C rows held by this lane

    float sumexp = 0.0f, l0 = 0.0f;

    for (int t = 0; t < 17; ++t) {         // 17 tiles x 16 rows >= 257
      const int cid = t * 16 + r;
      const float* rowp;
      if (cid == 0) {
        rowp = prow;                        // positive = prototype row c+1
      } else {
        int j = cid - 1; if (j > 255) j = 255;  // pad rows (ignored below)
        rowp = fore_rep + (long)nrow[j] * 128;
      }

      // A: raw row values -> f16; per-row scale applied post-WMMA in f32.
      // A k order used here (same as B): lanes 0-15 K{0-15}, lanes 16-31 K{16-31}.
      v16h A[4];
      float ssr = 0.0f;
#pragma unroll
      for (int ch = 0; ch < 4; ++ch) {
        const float* p0 = rowp + ch * 32 + kbase;
        const float* p1 = p0 + 8;
        float4 y0 = ((const float4*)p0)[0];
        float4 y1 = ((const float4*)p0)[1];
        float4 y2 = ((const float4*)p1)[0];
        float4 y3 = ((const float4*)p1)[1];
        ssr += y0.x*y0.x + y0.y*y0.y + y0.z*y0.z + y0.w*y0.w
             + y1.x*y1.x + y1.y*y1.y + y1.z*y1.z + y1.w*y1.w
             + y2.x*y2.x + y2.y*y2.y + y2.z*y2.z + y2.w*y2.w
             + y3.x*y3.x + y3.y*y3.y + y3.z*y3.z + y3.w*y3.w;
        A[ch][0]=(_Float16)y0.x;  A[ch][1]=(_Float16)y0.y;
        A[ch][2]=(_Float16)y0.z;  A[ch][3]=(_Float16)y0.w;
        A[ch][4]=(_Float16)y1.x;  A[ch][5]=(_Float16)y1.y;
        A[ch][6]=(_Float16)y1.z;  A[ch][7]=(_Float16)y1.w;
        A[ch][8]=(_Float16)y2.x;  A[ch][9]=(_Float16)y2.y;
        A[ch][10]=(_Float16)y2.z; A[ch][11]=(_Float16)y2.w;
        A[ch][12]=(_Float16)y3.x; A[ch][13]=(_Float16)y3.y;
        A[ch][14]=(_Float16)y3.z; A[ch][15]=(_Float16)y3.w;
      }
      // lanes r and r^16 hold disjoint halves of row r -> full sum of squares
      float ssf = ssr + __shfl_xor(ssr, 16, 32);
      float scl = fast_rsq(fmaxf(ssf, 1e-16f));

      v8f acc = {};
#pragma unroll
      for (int ch = 0; ch < 4; ++ch)
        acc = __builtin_amdgcn_wmma_f32_16x16x32_f16(
            false, A[ch], false, B[ch], (short)0, acc, false, false);

      // C layout: VGPR j of lane l holds M = (l<16?0:8)+j (all N identical).
#pragma unroll
      for (int j = 0; j < 8; ++j) {
        int   m  = mbase + j;
        float sm = __shfl(scl, m, 32);      // lane m holds row m's scale
        float lg = INV_T * acc[j] * sm;     // logit / TEMP
        int   cm = t * 16 + m;
        if (cm < 257) sumexp += fast_exp(lg); // |lg|<=2 -> no max-shift needed
        if (cm == 0)  l0 = lg;
      }
    }
    sumexp += __shfl_xor(sumexp, 16, 32);   // rows 0-7 half + rows 8-15 half
    l0 = __shfl(l0, 0, 32);
    float ce = fast_log(sumexp) - l0;       // -log_softmax[...,0]
    if (lane == 0) ws[pair] = ce;

  } else {
    // ============ BG: candidate-axis (axis=2) normalization ============
    // bg_logits[d] = sign-ish(a_d)/sqrt(257) * S_d / max(||col_d||, eps)
    float4 av = ((const float4*)arow)[lane];    // this lane's 4 dims
    const int* __restrict__ nrow = nbg + (long)pair * 256;

    float4 S, SS;
    {
      float4 v = ((const float4*)prow)[lane];
      S = v;
      SS.x = v.x*v.x; SS.y = v.y*v.y; SS.z = v.z*v.z; SS.w = v.w*v.w;
    }
    for (int j = 0; j < 256; j += 4) {
      int i0 = nrow[j], i1 = nrow[j+1], i2 = nrow[j+2], i3 = nrow[j+3];
      float4 v0 = ((const float4*)(memo + (long)i0 * 128))[lane];
      float4 v1 = ((const float4*)(memo + (long)i1 * 128))[lane];
      float4 v2 = ((const float4*)(memo + (long)i2 * 128))[lane];
      float4 v3 = ((const float4*)(memo + (long)i3 * 128))[lane];
      S.x  += v0.x + v1.x + v2.x + v3.x;
      S.y  += v0.y + v1.y + v2.y + v3.y;
      S.z  += v0.z + v1.z + v2.z + v3.z;
      S.w  += v0.w + v1.w + v2.w + v3.w;
      SS.x += v0.x*v0.x + v1.x*v1.x + v2.x*v2.x + v3.x*v3.x;
      SS.y += v0.y*v0.y + v1.y*v1.y + v2.y*v2.y + v3.y*v3.y;
      SS.z += v0.z*v0.z + v1.z*v1.z + v2.z*v2.z + v3.z*v3.z;
      SS.w += v0.w*v0.w + v1.w*v1.w + v2.w*v2.w + v3.w*v3.w;
    }

    float a4[4]  = {av.x, av.y, av.z, av.w};
    float Sv[4]  = {S.x,  S.y,  S.z,  S.w};
    float SSv[4] = {SS.x, SS.y, SS.z, SS.w};
    float sumexp = 0.0f, l0 = 0.0f;
#pragma unroll
    for (int e = 0; e < 4; ++e) {
      float an = a4[e] * fast_rcp(fmaxf(fabsf(a4[e]) * SQRT257, EPSF));
      float lg = INV_T * an * Sv[e] * fast_rsq(fmaxf(SSv[e], 1e-16f));
      sumexp += fast_exp(lg);               // |lg| <= 2 (Cauchy-Schwarz)
      if (lane == 0 && e == 0) l0 = lg;     // softmax index 0 = dim d=0
    }
    sumexp = wave_sum32(sumexp);            // over all 128 dims
    l0 = __shfl(l0, 0, 32);
    if (lane == 0) ws[2048 + pair] = fast_log(sumexp) - l0;
  }
}

// Deterministic fixed-order reduction of the 4096 CE values -> 2 losses.
__global__ __launch_bounds__(256) void reduce_kernel(const float* __restrict__ ws,
                                                     float* __restrict__ out)
{
  __shared__ float sf[256];
  __shared__ float sb[256];
  const int t = threadIdx.x;
  float f = 0.0f, b = 0.0f;
#pragma unroll
  for (int j = 0; j < 8; ++j) {
    f += ws[t + 256 * j];
    b += ws[2048 + t + 256 * j];
  }
  sf[t] = f; sb[t] = b;
  __syncthreads();
  for (int s = 128; s > 0; s >>= 1) {
    if (t < s) { sf[t] += sf[t + s]; sb[t] += sb[t + s]; }
    __syncthreads();
  }
  if (t == 0) {
    out[0] = sf[0] * (1.0f / 2048.0f);  // mean over q, sum over c, / C
    out[1] = sb[0] * (1.0f / 2048.0f);
  }
}

extern "C" void kernel_launch(void* const* d_in, const int* in_sizes, int n_in,
                              void* d_out, int out_size, void* d_ws, size_t ws_size,
                              hipStream_t stream) {
  const float* fore_rep   = (const float*)d_in[0];
  const float* proto      = (const float*)d_in[1];
  const float* memo       = (const float*)d_in[2];
  const int*   anchor_idx = (const int*)  d_in[3];
  const int*   nfg        = (const int*)  d_in[4];
  const int*   nbg        = (const int*)  d_in[5];
  float*       ws         = (float*)d_ws;
  float*       out        = (float*)d_out;

  // 4096 wave-tasks (2048 fg + 2048 bg), 8 waves (256 threads) per block.
  contrast_kernel<<<512, 256, 0, stream>>>(fore_rep, proto, memo,
                                           anchor_idx, nfg, nbg, ws);
  reduce_kernel<<<1, 256, 0, stream>>>(ws, out);
}
